// IntraBandBiMamba_28819230556258
// MI455X (gfx1250) — matrix operand order repaired and checked by reading, not compiled, not run on hardware
//
#include <hip/hip_runtime.h>
#include <math.h>

// ---------------------------------------------------------------------------
// IntraBandBiMamba for MI455X (gfx1250, wave32).
// GEMMs use v_wmma_f32_16x16x32_bf16 (f32 accumulate), N-blocked 4 tiles/wave.
// Scan is f32 VALU with LDS-staged B/C. Direction handled by index reversal.
// ---------------------------------------------------------------------------

typedef __attribute__((ext_vector_type(16))) __bf16 v16bf;
typedef __attribute__((ext_vector_type(8)))  float  v8f;

static constexpr int Bb = 2;
static constexpr int Nd = 128;   // d_model
static constexpr int Tt = 1024;
static constexpr int Kb = 32;
static constexpr int BK = Bb * Kb;          // 64 sequences
static constexpr int Mrows = BK * Tt;       // 65536 GEMM rows

__device__ __forceinline__ __bf16 f2bf(float f) { return (__bf16)f; }

// --------------------------- LayerNorm + transpose --------------------------
// x (B,N,T,K) -> xn (BK, T, N), LN over N. One block per (b,t).
__global__ void ln_transpose_kernel(const float* __restrict__ x,
                                    const float* __restrict__ g,
                                    const float* __restrict__ beta,
                                    float* __restrict__ xn) {
  __shared__ float slab[Kb * 132];   // [k][n], padded vs 64 banks
  __shared__ float smu[Kb], srs[Kb];
  const int bid = blockIdx.x;
  const int b = bid / Tt, t = bid % Tt;
  const int tid = threadIdx.x;

  for (int idx = tid; idx < Nd * Kb; idx += 256) {
    int n = idx >> 5, k = idx & 31;   // k fastest -> coalesced global read
    slab[k * 132 + n] = x[(((size_t)b * Nd + n) * Tt + t) * Kb + k];
  }
  __syncthreads();
  if (tid < Kb) {
    float s = 0.f, ss = 0.f;
    for (int n = 0; n < Nd; ++n) {
      float v = slab[tid * 132 + n];
      s += v; ss += v * v;
    }
    float mu = s * (1.f / Nd);
    float var = ss * (1.f / Nd) - mu * mu;
    smu[tid] = mu;
    srs[tid] = rsqrtf(var + 1e-5f);
  }
  __syncthreads();
  for (int idx = tid; idx < Nd * Kb; idx += 256) {
    int n = idx & (Nd - 1), k = idx >> 7;   // n fastest -> coalesced store
    float v = (slab[k * 132 + n] - smu[k]) * srs[k] * g[n] + beta[n];
    xn[(((size_t)(b * Kb + k)) * Tt + t) * Nd + n] = v;
  }
}

// --------------------------- generic WMMA GEMM ------------------------------
// C[M x Nreal] = A[M x K] * W[K x N]. One wave computes a 16 x (16*NBLK)
// strip: NBLK accumulators share one A fragment. GUARD handles Nreal not a
// multiple of 16 branch-free (clamped address + multiply-by-zero).
template <int NBLK, bool GUARD>
__global__ void wmma_gemm_kernel(const float* __restrict__ A,
                                 const float* __restrict__ W,
                                 float* __restrict__ C,
                                 int Mtiles, int Ngroups, int Nreal,
                                 int Kdim, int lda, int ldw, int ldc) {
  const int gw = (blockIdx.x * blockDim.x + threadIdx.x) >> 5;
  if (gw >= Mtiles * Ngroups) return;            // wave-uniform exit
  const int mt = gw / Ngroups, ng = gw % Ngroups;
  const int m0 = mt * 16;
  const int lane = threadIdx.x & 31;
  const int half = lane >> 4, ml = lane & 15;

  v8f c[NBLK];
#pragma unroll
  for (int q = 0; q < NBLK; ++q) c[q] = {};

  for (int k0 = 0; k0 < Kdim; k0 += 32) {
    v16bf a;
    const float* arow = A + (size_t)(m0 + ml) * lda + k0;
#pragma unroll
    for (int v = 0; v < 8; ++v) {
      // 16-bit A 16x32 layout (ISA 7.12.2): lanes 0-15 rows, pairs of K
      int kk = (v < 4) ? (8 * half + 2 * v) : (8 + 8 * half + 2 * v);
      float2 p = *(const float2*)(arow + kk);
      a[2 * v]     = f2bf(p.x);
      a[2 * v + 1] = f2bf(p.y);
    }
#pragma unroll
    for (int q = 0; q < NBLK; ++q) {
      int ncol = (ng * NBLK + q) * 16 + ml;
      float scale = 1.f;
      int nc = ncol;
      if (GUARD) {                     // branch-free: clamp addr, zero data
        if (ncol >= Nreal) { nc = Nreal - 1; scale = 0.f; }
      }
      v16bf w;
#pragma unroll
      for (int v = 0; v < 8; ++v) {
        // B 32x16: lane = column, K order matches A (8*half+s, 16+8*half+s-8)
        int kk = (v < 4) ? (8 * half + 2 * v) : (8 + 8 * half + 2 * v);
        w[2 * v]     = f2bf(W[(size_t)(k0 + kk) * ldw + nc] * scale);
        w[2 * v + 1] = f2bf(W[(size_t)(k0 + kk + 1) * ldw + nc] * scale);
      }
      c[q] = __builtin_amdgcn_wmma_f32_16x16x32_bf16(false, a, false, w,
                                                     (short)0, c[q], false, false);
    }
  }
#pragma unroll
  for (int q = 0; q < NBLK; ++q) {
    int ncol = (ng * NBLK + q) * 16 + ml;
    if (GUARD && ncol >= Nreal) continue;        // masked store only
#pragma unroll
    for (int r = 0; r < 8; ++r)   // C/D layout: row = r + 8*half, col = ml
      C[(size_t)(m0 + r + 8 * half) * ldc + ncol] = c[q][r];
  }
}

// --------------------------- depthwise conv + SiLU --------------------------
// xz (BK,T,256): columns [0,128) are xi. rev=0: causal; rev=1: anti-causal.
__global__ void conv_silu_kernel(const float* __restrict__ xz,
                                 const float* __restrict__ w,
                                 const float* __restrict__ bias,
                                 float* __restrict__ xc, int rev) {
  size_t idx = (size_t)blockIdx.x * blockDim.x + threadIdx.x;
  int d = idx & 127;
  int t = (int)((idx >> 7) & (Tt - 1));
  size_t bk = idx >> 17;
  float acc = bias[d];
#pragma unroll
  for (int j = 0; j < 4; ++j) {
    int tt = rev ? (t + 3 - j) : (t + j - 3);
    if (tt >= 0 && tt < Tt)
      acc += w[d * 4 + j] * xz[(bk * Tt + tt) * 256 + d];
  }
  xc[idx] = acc / (1.f + __expf(-acc));   // silu
}

// --------------------------- dt projection + softplus -----------------------
__global__ void dt_kernel(const float* __restrict__ dbl,
                          const float* __restrict__ dtw,
                          const float* __restrict__ dtb,
                          float* __restrict__ dt) {
  size_t idx = (size_t)blockIdx.x * blockDim.x + threadIdx.x;
  int d = idx & 127;
  size_t row = idx >> 7;   // bk*T + t
  float acc = dtb[d];
#pragma unroll
  for (int r = 0; r < 8; ++r)
    acc += dbl[row * 40 + r] * dtw[r * 128 + d];
  dt[idx] = (acc > 20.f) ? acc : log1pf(__expf(acc));
}

// --------------------------- selective scan + gate --------------------------
// grid (BK, 2), 128 threads: thread = channel d, 16 states in registers.
// B/C (shared per timestep across channels) staged through LDS in chunks.
// Writes g = (y + u*D)*silu(z) in-place over the dt buffer.
__global__ void scan_gate_kernel(
    float* __restrict__ dtg_f, float* __restrict__ dtg_b,
    const float* __restrict__ xc_f, const float* __restrict__ xc_b,
    const float* __restrict__ xz_f, const float* __restrict__ xz_b,
    const float* __restrict__ dbl_f, const float* __restrict__ dbl_b,
    const float* __restrict__ Al_f, const float* __restrict__ Al_b,
    const float* __restrict__ Dp_f, const float* __restrict__ Dp_b) {
  constexpr int CH = 64;
  const int dir = blockIdx.y;
  float* dtg       = dir ? dtg_b : dtg_f;
  const float* xc  = dir ? xc_b  : xc_f;
  const float* xz  = dir ? xz_b  : xz_f;
  const float* dbl = dir ? dbl_b : dbl_f;
  const float* Al  = dir ? Al_b  : Al_f;
  const float* Dpp = dir ? Dp_b  : Dp_f;
  const int bk = blockIdx.x;
  const int d = threadIdx.x;

  float A[16], h[16];
#pragma unroll
  for (int s = 0; s < 16; ++s) { A[s] = -__expf(Al[d * 16 + s]); h[s] = 0.f; }
  const float Dd = Dpp[d];

  __shared__ float sBC[CH][32];   // [step][0..15]=B, [16..31]=C
  for (int cc = 0; cc < Tt / CH; ++cc) {
    int t0 = dir ? (Tt - (cc + 1) * CH) : cc * CH;
    __syncthreads();
    for (int idx = threadIdx.x; idx < CH * 32; idx += blockDim.x) {
      int st = idx >> 5, o = idx & 31;
      sBC[st][o] = dbl[((size_t)bk * Tt + t0 + st) * 40 + 8 + o];
    }
    __syncthreads();
    for (int i = 0; i < CH; ++i) {
      int tl = dir ? (CH - 1 - i) : i;
      size_t ro = (size_t)bk * Tt + (t0 + tl);
      float dtv = dtg[ro * 128 + d];
      float u   = xc[ro * 128 + d];
      float zv  = xz[ro * 256 + 128 + d];
      float du = dtv * u;
      float y = 0.f;
#pragma unroll
      for (int s = 0; s < 16; ++s) {
        float dA = __expf(dtv * A[s]);
        h[s] = dA * h[s] + du * sBC[tl][s];
        y += h[s] * sBC[tl][16 + s];
      }
      float yd = y + u * Dd;
      float sz = zv / (1.f + __expf(-zv));
      dtg[ro * 128 + d] = yd * sz;    // safe: same thread, read-then-write
    }
  }
}

// --------------------------- combine GEMM + residual + scatter --------------
// out(b,n,t,k) = [out_f | out_b](row, :) @ combine_w + combine_b + x(b,n,t,k)
// One wave computes a 16x64 strip (4 accumulators share one A fragment).
__global__ void combine_kernel(const float* __restrict__ of,
                               const float* __restrict__ ob,
                               const float* __restrict__ W,
                               const float* __restrict__ bias,
                               const float* __restrict__ x,
                               float* __restrict__ out) {
  const int gw = (blockIdx.x * blockDim.x + threadIdx.x) >> 5;
  const int mt = gw >> 1, ng = gw & 1;          // Ngroups = 2 (128/64)
  const int m0 = mt * 16;
  const int lane = threadIdx.x & 31;
  const int half = lane >> 4, ml = lane & 15;

  v8f c[4];
#pragma unroll
  for (int q = 0; q < 4; ++q) c[q] = {};

  for (int k0 = 0; k0 < 256; k0 += 32) {
    const float* src = (k0 < 128) ? of : ob;
    const float* arow = src + (size_t)(m0 + ml) * 128 + (k0 & 127);
    v16bf a;
#pragma unroll
    for (int v = 0; v < 8; ++v) {
      int kk = (v < 4) ? (8 * half + 2 * v) : (8 + 8 * half + 2 * v);
      float2 p = *(const float2*)(arow + kk);
      a[2 * v]     = f2bf(p.x);
      a[2 * v + 1] = f2bf(p.y);
    }
#pragma unroll
    for (int q = 0; q < 4; ++q) {
      int n = (ng * 4 + q) * 16 + ml;
      v16bf w;
#pragma unroll
      for (int v = 0; v < 8; ++v) {
        int kk = (v < 4) ? (8 * half + 2 * v) : (8 + 8 * half + 2 * v);
        w[2 * v]     = f2bf(W[(size_t)(k0 + kk) * 128 + n]);
        w[2 * v + 1] = f2bf(W[(size_t)(k0 + kk + 1) * 128 + n]);
      }
      c[q] = __builtin_amdgcn_wmma_f32_16x16x32_bf16(false, a, false, w,
                                                     (short)0, c[q], false, false);
    }
  }
#pragma unroll
  for (int q = 0; q < 4; ++q) {
    int n = (ng * 4 + q) * 16 + ml;
    float bn = bias[n];
#pragma unroll
    for (int r = 0; r < 8; ++r) {
      int row = m0 + r + 8 * half;           // = bk*T + t
      int bk = row >> 10, t = row & (Tt - 1);
      int b = bk >> 5, kq = bk & 31;
      size_t base = (((size_t)b * Nd + n) * Tt + t) * Kb + kq;
      out[base] = c[q][r] + bn + x[base];    // fused residual (= xr)
    }
  }
}

// ---------------------------------------------------------------------------
extern "C" void kernel_launch(void* const* d_in, const int* in_sizes, int n_in,
                              void* d_out, int out_size, void* d_ws, size_t ws_size,
                              hipStream_t stream) {
  const float* x         = (const float*)d_in[0];
  const float* ln_g      = (const float*)d_in[1];
  const float* ln_b      = (const float*)d_in[2];
  const float* combine_w = (const float*)d_in[3];
  const float* combine_b = (const float*)d_in[4];
  const float* f_in_w    = (const float*)d_in[5];
  const float* f_conv_w  = (const float*)d_in[6];
  const float* f_conv_b  = (const float*)d_in[7];
  const float* f_xproj_w = (const float*)d_in[8];
  const float* f_dt_w    = (const float*)d_in[9];
  const float* f_dt_b    = (const float*)d_in[10];
  const float* f_A_log   = (const float*)d_in[11];
  const float* f_D       = (const float*)d_in[12];
  const float* f_out_w   = (const float*)d_in[13];
  const float* b_in_w    = (const float*)d_in[14];
  const float* b_conv_w  = (const float*)d_in[15];
  const float* b_conv_b  = (const float*)d_in[16];
  const float* b_xproj_w = (const float*)d_in[17];
  const float* b_dt_w    = (const float*)d_in[18];
  const float* b_dt_b    = (const float*)d_in[19];
  const float* b_A_log   = (const float*)d_in[20];
  const float* b_D       = (const float*)d_in[21];
  const float* b_out_w   = (const float*)d_in[22];
  float* out = (float*)d_out;

  float* ws = (float*)d_ws;
  const size_t S = (size_t)BK * Tt * Nd;        // 8M floats
  float* xn    = ws;                             // S
  float* xz_f  = ws + S;                         // 2S  (stride 256: xi|z)
  float* xz_b  = ws + 3 * S;                     // 2S
  float* xc_f  = ws + 5 * S;                     // S   (later reused as out_f)
  float* xc_b  = ws + 6 * S;                     // S   (later reused as out_b)
  float* dt_f  = ws + 7 * S;                     // S   (later reused as gate g)
  float* dt_b  = ws + 8 * S;                     // S
  float* dbl_f = ws + 9 * S;                     // BK*T*40
  float* dbl_b = dbl_f + (size_t)BK * Tt * 40;

  const int ELW = (int)(S / 256);                // 32768 blocks of 256 threads
  const int Mtiles = Mrows / 16;                 // 4096

  // 1. LayerNorm + transpose
  ln_transpose_kernel<<<Bb * Tt, 256, 0, stream>>>(x, ln_g, ln_b, xn);

  // 2. in-proj (M=65536, N=256, K=128): 16 n-tiles -> 4 groups of 4
  {
    int waves = Mtiles * 4, blocks = (waves + 7) / 8;
    wmma_gemm_kernel<4, false><<<blocks, 256, 0, stream>>>(xn, f_in_w, xz_f,
        Mtiles, 4, 256, 128, Nd, 256, 256);
    wmma_gemm_kernel<4, false><<<blocks, 256, 0, stream>>>(xn, b_in_w, xz_b,
        Mtiles, 4, 256, 128, Nd, 256, 256);
  }

  // 3. depthwise conv + SiLU (backward = anti-causal window)
  conv_silu_kernel<<<ELW, 256, 0, stream>>>(xz_f, f_conv_w, f_conv_b, xc_f, 0);
  conv_silu_kernel<<<ELW, 256, 0, stream>>>(xz_b, b_conv_w, b_conv_b, xc_b, 1);

  // 4. x-proj (N=40 -> one guarded 3-tile group per wave)
  {
    int waves = Mtiles, blocks = (waves + 7) / 8;
    wmma_gemm_kernel<3, true><<<blocks, 256, 0, stream>>>(xc_f, f_xproj_w, dbl_f,
        Mtiles, 1, 40, 128, Nd, 40, 40);
    wmma_gemm_kernel<3, true><<<blocks, 256, 0, stream>>>(xc_b, b_xproj_w, dbl_b,
        Mtiles, 1, 40, 128, Nd, 40, 40);
  }

  // 5. dt = softplus(dbl[:, :8] @ dt_w + dt_b)   (K=8 -> VALU)
  dt_kernel<<<ELW, 256, 0, stream>>>(dbl_f, f_dt_w, f_dt_b, dt_f);
  dt_kernel<<<ELW, 256, 0, stream>>>(dbl_b, b_dt_w, b_dt_b, dt_b);

  // 6. selective scan + gate (both directions in one launch; g overwrites dt)
  {
    dim3 grid(BK, 2);
    scan_gate_kernel<<<grid, 128, 0, stream>>>(dt_f, dt_b, xc_f, xc_b,
        xz_f, xz_b, dbl_f, dbl_b, f_A_log, b_A_log, f_D, b_D);
  }

  // 7. out-proj (N=128, K=128): 8 n-tiles -> 2 groups of 4
  {
    int waves = Mtiles * 2, blocks = (waves + 7) / 8;
    wmma_gemm_kernel<4, false><<<blocks, 256, 0, stream>>>(dt_f, f_out_w, xc_f,
        Mtiles, 2, 128, 128, Nd, 128, 128);
    wmma_gemm_kernel<4, false><<<blocks, 256, 0, stream>>>(dt_b, b_out_w, xc_b,
        Mtiles, 2, 128, 128, Nd, 128, 128);
  }

  // 8. combine (K=256) + bias + residual + scatter to (B,N,T,K)
  {
    int waves = Mtiles * 2, blocks = (waves + 7) / 8;   // 1024 blocks exact
    combine_kernel<<<blocks, 256, 0, stream>>>(xc_f, xc_b, combine_w,
        combine_b, x, out);
  }
}
